// FlashMoeBlockWrapper_3624952398281
// MI455X (gfx1250) — matrix operand order repaired and checked by reading, not compiled, not run on hardware
//
#include <hip/hip_runtime.h>
#include <stdint.h>

// ---------------------------------------------------------------------------
// MoE block (Qwen3-style) for gfx1250: routed top-8, f16 WMMA expert GEMMs.
// T=1024, D=2048, E=64, F=768, top-k=8. Deterministic (no atomics).
// GEMMs are software-pipelined: tile k+1 staged in VGPRs during tile k's WMMAs.
// ---------------------------------------------------------------------------

typedef __attribute__((ext_vector_type(16))) _Float16 v16h;
typedef __attribute__((ext_vector_type(8)))  float    v8f;

#define T_TOK 1024
#define DIM   2048
#define NEXP  64
#define FDIM  768
#define TOPK  8
#define NTOT  (T_TOK * TOPK)          // 8192 (token,slot) entries

union FragH { v16h h; unsigned int u[8]; };
union Acc8  { v8f  v; float f[8]; };

__device__ __forceinline__ unsigned short f2h(float f) {
  _Float16 h = (_Float16)f;
  return __builtin_bit_cast(unsigned short, h);
}

__device__ __forceinline__ v8f zero8() {
  v8f z = {0.f, 0.f, 0.f, 0.f, 0.f, 0.f, 0.f, 0.f};
  return z;
}

// A-fragment (16x32 f16, ISA 7.12.2): lane -> M = lane&15, hi = lane>>4.
// VGPR v holds K halves k0,k0+1 with k0 = (v&3)*2 + (v>>2)*16 + hi*8.
// LDS tile is row-major, K-contiguous, row stride 34 halves (dword aligned).
__device__ __forceinline__ v16h ldsFragA(const unsigned short* base, int row, int hi) {
  FragH f;
  const unsigned short* r = base + row * 34;
#pragma unroll
  for (int v = 0; v < 8; ++v) {
    int k0 = ((v & 3) << 1) + ((v >> 2) << 4) + (hi << 3);
    f.u[v] = *(const unsigned int*)(r + k0);
  }
  return f.h;
}

// B-fragment (32x16 f16): column n = lane&15; same K order as A fragment:
// VGPR v: k0 = (v&3)*2 + (v>>2)*16 + hi*8.
// LDS tile stored N-major with K contiguous (weights are naturally this way).
__device__ __forceinline__ v16h ldsFragB(const unsigned short* base, int row, int hi) {
  FragH f;
  const unsigned short* r = base + row * 34;
#pragma unroll
  for (int v = 0; v < 8; ++v) {
    int k0 = ((v & 3) << 1) + ((v >> 2) << 4) + (hi << 3);
    f.u[v] = *(const unsigned int*)(r + k0);
  }
  return f.h;
}

__device__ __forceinline__ v8f wmma_f16(v16h a, v16h b, v8f c) {
  return __builtin_amdgcn_wmma_f32_16x16x32_f16(false, a, false, b, (short)0, c,
                                                false, false);
}

// ---------------------------------------------------------------------------
// Kernel 1: router. One block per token. logits = x @ gate_w^T, top-8,
// renormalized softmax over the selected logits.
// ---------------------------------------------------------------------------
__global__ __launch_bounds__(256) void moe_router(
    const float* __restrict__ x, const float* __restrict__ gw,
    int* __restrict__ topi, float* __restrict__ topw) {
  __shared__ float xs[DIM];
  __shared__ float part[NEXP][4];
  __shared__ float ls[NEXP];
  const int t = blockIdx.x;
  const int tid = threadIdx.x;

  // stage x row (8KB) in LDS
  for (int i = 0; i < 2; ++i) {
    int d = (tid + i * 256) * 4;
    *(float4*)&xs[d] = *(const float4*)(x + (size_t)t * DIM + d);
  }
  __syncthreads();

  // 4 threads per expert, each over a contiguous 512-float chunk
  const int e = tid >> 2, p = tid & 3;
  const float* g = gw + (size_t)e * DIM + p * 512;
  const float* xp = xs + p * 512;
  float s = 0.f;
  for (int i = 0; i < 512; i += 4) {
    float4 v = *(const float4*)(g + i);
    s += v.x * xp[i] + v.y * xp[i + 1] + v.z * xp[i + 2] + v.w * xp[i + 3];
  }
  part[e][p] = s;
  __syncthreads();
  if (tid < NEXP) ls[tid] = part[tid][0] + part[tid][1] + part[tid][2] + part[tid][3];
  __syncthreads();

  if (tid == 0) {
    unsigned long long taken = 0ull;
    float vals[TOPK];
    int idx[TOPK];
    for (int k = 0; k < TOPK; ++k) {
      float best = -3.0e38f;
      int bi = 0;
      for (int j = 0; j < NEXP; ++j)
        if (!((taken >> j) & 1ull) && ls[j] > best) { best = ls[j]; bi = j; }
      taken |= 1ull << bi;
      vals[k] = best;
      idx[k] = bi;
    }
    const float mx = vals[0];  // top-1 is the max logit
    float sum = 0.f;
    for (int k = 0; k < TOPK; ++k) { vals[k] = __expf(vals[k] - mx); sum += vals[k]; }
    const float inv = 1.0f / sum;
    for (int k = 0; k < TOPK; ++k) {
      topi[t * TOPK + k] = idx[k];
      topw[t * TOPK + k] = vals[k] * inv;
    }
  }
}

// ---------------------------------------------------------------------------
// Kernel 2: per-expert entry lists. One wave per expert; deterministic order
// via ballot + prefix popcount. Entry id = token*8 + slot.
// ---------------------------------------------------------------------------
__global__ __launch_bounds__(32) void moe_build(
    const int* __restrict__ topi, int* __restrict__ list, int* __restrict__ cnt) {
  const int e = blockIdx.x;
  const int lane = threadIdx.x;
  int base = 0;
  for (int i0 = 0; i0 < NTOT; i0 += 32) {
    const int v = topi[i0 + lane];
    const bool m = (v == e);
    const unsigned mask = __builtin_amdgcn_ballot_w32(m);
    if (m) {
      const int pos = base + __builtin_popcount(mask & ((1u << lane) - 1u));
      list[e * T_TOK + pos] = i0 + lane;
    }
    base += __builtin_popcount(mask);
  }
  if (lane == 0) cnt[e] = base;
}

// ---------------------------------------------------------------------------
// Kernel 3: expert up/gate GEMM + fused SwiGLU -> h (f16).
// grid = (6, 64): 6 column blocks of (128 gate + 128 up) cols, one expert per y.
// Block tile 32M x 256N x 32K; 8 waves = 2M x 4N; each wave 16M x 64N
// (32 gate + 32 up cols so SwiGLU pairs meet in the same lane/VGPR slot).
// Software pipelined: global fp32 tile k+1 staged in VGPRs while tile k's
// WMMAs run; conversion to f16 happens at LDS-store time.
// ---------------------------------------------------------------------------
__global__ __launch_bounds__(256) void moe_gemm1(
    const float* __restrict__ x, const float* __restrict__ w13,
    const int* __restrict__ list, const int* __restrict__ cnt,
    _Float16* __restrict__ hbuf) {
  const int e = blockIdx.y;
  const int Ne = cnt[e];
  if (Ne == 0) return;
  const int n0 = blockIdx.x * 128;  // gate col base (up cols at +FDIM)

  __shared__ unsigned short Ash[32 * 34];
  __shared__ unsigned short Bsh[256 * 34];

  const int tid = threadIdx.x;
  const int lane = tid & 31;
  const int wid = tid >> 5;
  const int mwoff = (wid & 1) * 16;
  const int nw = wid >> 1;  // 0..3
  const int l15 = lane & 15;
  const int hi = lane >> 4;
  const int* elist = list + e * T_TOK;

  // B loader mapping: row r=tid -> (rnw = r>>6, q = r&63):
  //   q<32: gate col n0+rnw*32+q ; q>=32: up col FDIM+n0+rnw*32+(q-32)
  const int q = tid & 63, rnw = tid >> 6;
  const int bcol = (q < 32) ? (n0 + rnw * 32 + q) : (FDIM + n0 + rnw * 32 + (q - 32));
  const float* wrow = w13 + ((size_t)e * (2 * FDIM) + bcol) * DIM;

  const int mtiles = (Ne + 31) / 32;
  for (int mt = 0; mt < mtiles; ++mt) {
    const int mbase = mt * 32;
    v8f a0 = zero8(), a1 = zero8(), a2 = zero8(), a3 = zero8();

    // A loader mapping: row am = tid>>3, halves kk=(tid&7)*4
    const int am = tid >> 3;
    const int arow = mbase + am;
    const bool av = (arow < Ne);
    const int tok = av ? (elist[arow] >> 3) : 0;
    const int kk = (tid & 7) * 4;
    const float* xrow = x + (size_t)tok * DIM + kk;

    // prologue: stage tile k0=0 in registers
    float4 ax = make_float4(0.f, 0.f, 0.f, 0.f);
    float4 bx[8];
    if (av) ax = *(const float4*)(xrow);
#pragma unroll
    for (int i = 0; i < 8; ++i) bx[i] = *(const float4*)(wrow + i * 4);

    for (int k0 = 0; k0 < DIM; k0 += 32) {
      __syncthreads();  // previous iteration's fragment reads are done
      {  // convert staged fp32 -> f16 and store to LDS
        unsigned short* da = Ash + am * 34 + kk;
        da[0] = f2h(ax.x); da[1] = f2h(ax.y); da[2] = f2h(ax.z); da[3] = f2h(ax.w);
        unsigned short* db = Bsh + tid * 34;
#pragma unroll
        for (int i = 0; i < 8; ++i) {
          db[i * 4 + 0] = f2h(bx[i].x); db[i * 4 + 1] = f2h(bx[i].y);
          db[i * 4 + 2] = f2h(bx[i].z); db[i * 4 + 3] = f2h(bx[i].w);
        }
      }
      __syncthreads();
      // issue next tile's global loads; they overlap the WMMAs below
      const int kn = k0 + 32;
      if (kn < DIM) {
        if (av) ax = *(const float4*)(xrow + kn);
        const float* wp = wrow + kn;
#pragma unroll
        for (int i = 0; i < 8; ++i) bx[i] = *(const float4*)(wp + i * 4);
      }
      const v16h a = ldsFragA(Ash, mwoff + l15, hi);
      const v16h b0 = ldsFragB(Bsh, nw * 64 + l15, hi);
      const v16h b1 = ldsFragB(Bsh, nw * 64 + 16 + l15, hi);
      const v16h b2 = ldsFragB(Bsh, nw * 64 + 32 + l15, hi);
      const v16h b3 = ldsFragB(Bsh, nw * 64 + 48 + l15, hi);
      a0 = wmma_f16(a, b0, a0);
      a1 = wmma_f16(a, b1, a1);
      a2 = wmma_f16(a, b2, a2);
      a3 = wmma_f16(a, b3, a3);
    }

    // Epilogue: SwiGLU on (gate,up) register pairs, write h as f16.
    Acc8 c0, c1, c2, c3;
    c0.v = a0; c1.v = a1; c2.v = a2; c3.v = a3;
#pragma unroll
    for (int i = 0; i < 8; ++i) {
      const int M = mwoff + i + hi * 8;  // C/D layout: VGPR i -> M = i (+8 hi half)
      const int row = mbase + M;
      if (row < Ne) {
        const int entry = elist[row];
        const int n = n0 + nw * 32 + l15;
        const float g0 = c0.f[i], u0 = c2.f[i];
        const float g1 = c1.f[i], u1 = c3.f[i];
        const float h0 = (g0 / (1.0f + __expf(-g0))) * u0;
        const float h1 = (g1 / (1.0f + __expf(-g1))) * u1;
        _Float16* hp = hbuf + (size_t)entry * FDIM + n;
        hp[0] = (_Float16)h0;
        hp[16] = (_Float16)h1;
      }
    }
  }
}

// ---------------------------------------------------------------------------
// Kernel 4: expert down GEMM. h[Ne,768](f16) x w2_e^T -> weighted rows into
// contrib[entry][0..2047]. grid = (8, 64): 256 output cols per block.
// Same software pipelining as gemm1; A tile is already f16 in hbuf.
// ---------------------------------------------------------------------------
__global__ __launch_bounds__(256) void moe_gemm2(
    const _Float16* __restrict__ hbuf, const float* __restrict__ w2,
    const int* __restrict__ list, const int* __restrict__ cnt,
    const float* __restrict__ topw, float* __restrict__ contrib) {
  const int e = blockIdx.y;
  const int Ne = cnt[e];
  if (Ne == 0) return;
  const int n0 = blockIdx.x * 256;

  __shared__ unsigned short Ash[32 * 34];
  __shared__ unsigned short Bsh[256 * 34];

  const int tid = threadIdx.x;
  const int lane = tid & 31;
  const int wid = tid >> 5;
  const int mwoff = (wid & 1) * 16;
  const int nw = wid >> 1;  // 0..3, 64 cols each
  const int l15 = lane & 15;
  const int hi = lane >> 4;
  const int* elist = list + e * T_TOK;

  const float* wrow = w2 + ((size_t)e * DIM + (n0 + tid)) * FDIM;

  const int mtiles = (Ne + 31) / 32;
  for (int mt = 0; mt < mtiles; ++mt) {
    const int mbase = mt * 32;
    v8f a0 = zero8(), a1 = zero8(), a2 = zero8(), a3 = zero8();

    const int am = tid >> 3;
    const int arow = mbase + am;
    const bool av = (arow < Ne);
    const int entryA = av ? elist[arow] : 0;
    const int kk = (tid & 7) * 4;  // halves
    const _Float16* hrow = hbuf + (size_t)entryA * FDIM + kk;

    // prologue: stage tile k0=0 in registers
    uint2 axh = make_uint2(0u, 0u);
    float4 bx[8];
    if (av) axh = *(const uint2*)(hrow);
#pragma unroll
    for (int i = 0; i < 8; ++i) bx[i] = *(const float4*)(wrow + i * 4);

    for (int k0 = 0; k0 < FDIM; k0 += 32) {
      __syncthreads();
      {  // store staged tile to LDS (A already f16; B fp32 -> f16)
        *(unsigned int*)(Ash + am * 34 + kk) = axh.x;
        *(unsigned int*)(Ash + am * 34 + kk + 2) = axh.y;
        unsigned short* db = Bsh + tid * 34;
#pragma unroll
        for (int i = 0; i < 8; ++i) {
          db[i * 4 + 0] = f2h(bx[i].x); db[i * 4 + 1] = f2h(bx[i].y);
          db[i * 4 + 2] = f2h(bx[i].z); db[i * 4 + 3] = f2h(bx[i].w);
        }
      }
      __syncthreads();
      // issue next tile's global loads; they overlap the WMMAs below
      const int kn = k0 + 32;
      if (kn < FDIM) {
        if (av) axh = *(const uint2*)(hrow + kn);
        const float* wp = wrow + kn;
#pragma unroll
        for (int i = 0; i < 8; ++i) bx[i] = *(const float4*)(wp + i * 4);
      }
      const v16h a = ldsFragA(Ash, mwoff + l15, hi);
      const v16h b0 = ldsFragB(Bsh, nw * 64 + l15, hi);
      const v16h b1 = ldsFragB(Bsh, nw * 64 + 16 + l15, hi);
      const v16h b2 = ldsFragB(Bsh, nw * 64 + 32 + l15, hi);
      const v16h b3 = ldsFragB(Bsh, nw * 64 + 48 + l15, hi);
      a0 = wmma_f16(a, b0, a0);
      a1 = wmma_f16(a, b1, a1);
      a2 = wmma_f16(a, b2, a2);
      a3 = wmma_f16(a, b3, a3);
    }

    Acc8 c0, c1, c2, c3;
    c0.v = a0; c1.v = a1; c2.v = a2; c3.v = a3;
#pragma unroll
    for (int i = 0; i < 8; ++i) {
      const int M = mwoff + i + hi * 8;
      const int row = mbase + M;
      if (row < Ne) {
        const int entry = elist[row];
        const float w = topw[entry];
        float* cp = contrib + (size_t)entry * DIM + n0 + nw * 64 + l15;
        cp[0]  = w * c0.f[i];
        cp[16] = w * c1.f[i];
        cp[32] = w * c2.f[i];
        cp[48] = w * c3.f[i];
      }
    }
  }
}

// ---------------------------------------------------------------------------
// Kernel 5: out[t][d] = sum over the token's 8 slots (fixed order: determinism)
// ---------------------------------------------------------------------------
__global__ __launch_bounds__(256) void moe_reduce(
    const float* __restrict__ contrib, float* __restrict__ out) {
  const int t = blockIdx.x;
  const int tid = threadIdx.x;
#pragma unroll
  for (int i = 0; i < 2; ++i) {
    const int d = (tid + i * 256) * 4;
    float4 s = make_float4(0.f, 0.f, 0.f, 0.f);
#pragma unroll
    for (int k = 0; k < TOPK; ++k) {
      const float4 v = *(const float4*)(contrib + ((size_t)t * TOPK + k) * DIM + d);
      s.x += v.x; s.y += v.y; s.z += v.z; s.w += v.w;
    }
    *(float4*)(out + (size_t)t * DIM + d) = s;
  }
}

// ---------------------------------------------------------------------------
extern "C" void kernel_launch(void* const* d_in, const int* in_sizes, int n_in,
                              void* d_out, int out_size, void* d_ws, size_t ws_size,
                              hipStream_t stream) {
  const float* x   = (const float*)d_in[0];   // [1024, 2048]
  const float* gw  = (const float*)d_in[1];   // [64, 2048]
  const float* w13 = (const float*)d_in[2];   // [64, 1536, 2048]
  const float* w2  = (const float*)d_in[3];   // [64, 2048, 768]
  float* out = (float*)d_out;                 // [1024, 2048]

  uint8_t* ws = (uint8_t*)d_ws;
  int*      topi    = (int*)(ws);                       // 8192 ints   (32KB)
  float*    topw    = (float*)(ws + (32u << 10));       // 8192 floats (32KB)
  int*      cnt     = (int*)(ws + (64u << 10));         // 64 ints
  int*      list    = (int*)(ws + (68u << 10));         // 64*1024 ints (256KB)
  _Float16* hbuf    = (_Float16*)(ws + (1u << 20));     // 8192*768 f16 (12MB)
  float*    contrib = (float*)(ws + (16u << 20));       // 8192*2048 f32 (64MB)

  moe_router<<<dim3(T_TOK), dim3(256), 0, stream>>>(x, gw, topi, topw);
  moe_build<<<dim3(NEXP), dim3(32), 0, stream>>>(topi, list, cnt);
  moe_gemm1<<<dim3(6, NEXP), dim3(256), 0, stream>>>(x, w13, list, cnt, hbuf);
  moe_gemm2<<<dim3(8, NEXP), dim3(256), 0, stream>>>(hbuf, w2, list, cnt, topw, contrib);
  moe_reduce<<<dim3(T_TOK), dim3(256), 0, stream>>>(contrib, out);
}
